// FormulaDecoder_89996744720551
// MI455X (gfx1250) — matrix operand order repaired and checked by reading, not compiled, not run on hardware
//
#include <hip/hip_runtime.h>

#define HID   1024
#define VOCAB 32000
#define BATCH 64
#define TLEN  128
#define KC    (HID / 32)          // K-chunks of 32 per GEMM K-loop

typedef __attribute__((ext_vector_type(16))) __bf16         v16bf;
typedef __attribute__((ext_vector_type(8)))  float          v8f;
typedef __attribute__((ext_vector_type(8)))  unsigned short u16x8;

union FragAB { v16bf bf; u16x8 u[2]; };

__device__ __forceinline__ unsigned short f2bf(float f) {
    // round-to-nearest-even float32 -> bf16
    unsigned int u = __float_as_uint(f);
    unsigned int r = u + 0x7FFFu + ((u >> 16) & 1u);
    return (unsigned short)(r >> 16);
}

// ---------------------------------------------------------------------------
// Packed fragment layouts (per-lane contiguous 32 B, per-fragment contiguous 1 KB)
//   fragment index (tile, kchunk) -> 512 u16 at offset ((tile*KC + kc)*512)
//   within a fragment: lane*16 + j   (j = position in the lane's 16 bf16)
// B-matrix (W row-major N x K, B = W^T):  lane = hi*16 + (n%16),
//   element j = W[n, kc*32 + 8*hi + (j&7) + 16*(j>>3)]
// A-matrix (M x K):                       lane = hi*16 + (m%16),
//   r = k%32: hi = (r>>3)&1, j = (r&7) + 8*((r>>4)&1)
// These mirror the ISA 7.12.2 16-bit A(16x32) / B(32x16) VGPR layouts exactly.
// ---------------------------------------------------------------------------

__device__ __forceinline__ long long a_pk_idx(int m, int k) {
    int mt = m >> 4, l = m & 15;
    int kc = k >> 5, r = k & 31;
    int hi = (r >> 3) & 1;
    int j  = (r & 7) + (((r >> 4) & 1) << 3);
    return ((long long)(mt * KC + kc) * 32 + hi * 16 + l) * 16 + j;
}

// ---------------- pack weights into B-fragment order (once per launch) -----
__global__ void pack_w_bf16(const float* __restrict__ src,
                            unsigned short* __restrict__ dst, long long total) {
    long long i = (long long)blockIdx.x * blockDim.x + threadIdx.x;  // dst-linear
    if (i >= total) return;
    int       j     = (int)(i & 15);
    int       lane  = (int)((i >> 4) & 31);
    long long chunk = i >> 9;              // nt*KC + kc
    int       kc    = (int)(chunk % KC);
    int       nt    = (int)(chunk / KC);
    int       hi    = lane >> 4, l = lane & 15;
    int       n     = nt * 16 + l;
    int       k     = kc * 32 + 8 * hi + (j & 7) + 16 * (j >> 3);
    dst[i] = f2bf(src[(long long)n * HID + k]);
}

// ---------------- state init (h fp32 + packed bf16, first tokens) ----------
__global__ void init_state(const float* __restrict__ enc, const int* __restrict__ tgt,
                           float* __restrict__ h, unsigned short* __restrict__ hpk,
                           int* __restrict__ tok) {
    int i = blockIdx.x * blockDim.x + threadIdx.x;
    if (i < BATCH * HID) {
        int m = i >> 10, k = i & 1023;
        float v = enc[i];
        h[i] = v;
        hpk[a_pk_idx(m, k)] = f2bf(v);
    }
    if (i < BATCH) tok[i] = tgt[i * TLEN];     // target_tokens[:, 0]
}

// ---------------- per-step embedding gather -> packed A fragments ----------
__global__ void embed_cast(const float* __restrict__ emb, const int* __restrict__ tok,
                           unsigned short* __restrict__ xpk) {
    int i = blockIdx.x * blockDim.x + threadIdx.x;   // 64*1024 threads
    int b = i >> 10, k = i & 1023;
    xpk[a_pk_idx(b, k)] = f2bf(emb[(long long)tok[b] * HID + k]);
}

// ---------------- WMMA GEMM wave tile: 32(M) x 64(N), K = HID --------------
__device__ __forceinline__ FragAB load_frag(const unsigned short* __restrict__ p,
                                            int lane) {
    FragAB f;
    f.u[0] = *(const u16x8*)(p + lane * 16);
    f.u[1] = *(const u16x8*)(p + lane * 16 + 8);
    return f;
}

__device__ __forceinline__ v8f wmma_bf16(const FragAB& a, const FragAB& b, v8f c) {
    return __builtin_amdgcn_wmma_f32_16x16x32_bf16(
        false, a.bf, false, b.bf, (short)0, c, false, false);
}

__device__ __forceinline__ void gemm_tile_32x64(
        const unsigned short* __restrict__ Apk,   // packed A fragments
        const unsigned short* __restrict__ Wpk,   // packed B fragments
        float* __restrict__ C, long long ldc,
        int mt0, int nt0, const float* __restrict__ bias) {
    int lane = threadIdx.x & 31;
    v8f acc[2][4] = {};

    const unsigned short* a0p = Apk + (long long)(mt0 + 0) * KC * 512;
    const unsigned short* a1p = Apk + (long long)(mt0 + 1) * KC * 512;

    for (int kc = 0; kc < KC; ++kc) {
        FragAB a0 = load_frag(a0p + (long long)kc * 512, lane);
        FragAB a1 = load_frag(a1p + (long long)kc * 512, lane);
#pragma unroll
        for (int j = 0; j < 4; ++j) {
            FragAB b = load_frag(Wpk + ((long long)(nt0 + j) * KC + kc) * 512, lane);
            acc[0][j] = wmma_bf16(a0, b, acc[0][j]);
            acc[1][j] = wmma_bf16(a1, b, acc[1][j]);
        }
    }

    int l = lane & 15, hi = lane >> 4;
#pragma unroll
    for (int i = 0; i < 2; ++i) {
#pragma unroll
        for (int j = 0; j < 4; ++j) {
            int   col = (nt0 + j) * 16 + l;
            float bv  = bias ? bias[col] : 0.0f;
#pragma unroll
            for (int r = 0; r < 8; ++r) {
                // C/D layout: VGPR r -> M = r (lanes 0-15) / 8+r (lanes 16-31)
                int m = (mt0 + i) * 16 + r + 8 * hi;
                C[(long long)m * ldc + col] = acc[i][j][r] + bv;
            }
        }
    }
}

// ---------------- GRU gate GEMMs: gi = x@w_ih^T, gh = h@w_hh^T -------------
__global__ void __launch_bounds__(128) gates_gemm(
        const unsigned short* __restrict__ xpk, const unsigned short* __restrict__ hpk,
        const unsigned short* __restrict__ wihp, const unsigned short* __restrict__ whhp,
        float* __restrict__ gi, float* __restrict__ gh) {
    int wave  = threadIdx.x >> 5;
    int strip = blockIdx.x * 4 + wave;          // 0..95  (48 N-strips * 2 matrices)
    int mt0   = blockIdx.y * 2;                 // 2 M-strips of 32 cover batch 64
    int isH   = strip >= 48;                    // wave-uniform select
    int nt0   = (strip - (isH ? 48 : 0)) * 4;   // 48 * 64 = 3072
    const unsigned short* A = isH ? hpk  : xpk;
    const unsigned short* W = isH ? whhp : wihp;
    float*                Cm = isH ? gh  : gi;
    gemm_tile_32x64(A, W, Cm, 3 * HID, mt0, nt0, nullptr);
}

// ---------------- GRU gate fusion ------------------------------------------
__global__ void gru_update(const float* __restrict__ gi, const float* __restrict__ gh,
                           const float* __restrict__ bih, const float* __restrict__ bhh,
                           float* __restrict__ h, unsigned short* __restrict__ hpk) {
    int i = blockIdx.x * blockDim.x + threadIdx.x;   // 64*1024 threads
    int b = i >> 10, j = i & 1023;
    const float* gib = gi + (long long)b * 3 * HID;
    const float* ghb = gh + (long long)b * 3 * HID;
    float ir = gib[j]           + bih[j],           hr = ghb[j]           + bhh[j];
    float iz = gib[HID + j]     + bih[HID + j],     hz = ghb[HID + j]     + bhh[HID + j];
    float in = gib[2 * HID + j] + bih[2 * HID + j], hn = ghb[2 * HID + j] + bhh[2 * HID + j];
    float r = 1.0f / (1.0f + expf(-(ir + hr)));
    float z = 1.0f / (1.0f + expf(-(iz + hz)));
    float n = tanhf(in + r * hn);
    float hv = (1.0f - z) * n + z * h[i];
    h[i] = hv;
    hpk[a_pk_idx(b, j)] = f2bf(hv);
}

// ---------------- logits GEMM: writes d_out slice directly -----------------
__global__ void __launch_bounds__(128) logits_gemm(
        const unsigned short* __restrict__ hpk, const unsigned short* __restrict__ fcwp,
        const float* __restrict__ fcb, float* __restrict__ out_t) {
    int wave  = threadIdx.x >> 5;
    int strip = blockIdx.x * 4 + wave;   // 0..499 ; 500 * 64 = 32000
    int mt0   = blockIdx.y * 2;
    int nt0   = strip * 4;
    // out layout: (B, T, V); out_t points at [0, t, 0]; row stride = T*V
    gemm_tile_32x64(hpk, fcwp, out_t, (long long)TLEN * VOCAB, mt0, nt0, fcb);
}

// ---------------- greedy argmax per batch row (first-max tie-break) --------
__global__ void argmax_rows(const float* __restrict__ out_t, int* __restrict__ tok) {
    __shared__ float sv[256];
    __shared__ int   si[256];
    int b = blockIdx.x, t = threadIdx.x;
    const float* row = out_t + (long long)b * TLEN * VOCAB;
    float best = -__builtin_inff();
    int   bi   = 0;
    for (int i = t; i < VOCAB; i += 256) {
        float v = row[i];
        if (v > best) { best = v; bi = i; }   // strict '>' keeps first index
    }
    sv[t] = best; si[t] = bi;
    __syncthreads();
    for (int s = 128; s > 0; s >>= 1) {
        if (t < s) {
            if (sv[t + s] > sv[t] || (sv[t + s] == sv[t] && si[t + s] < si[t])) {
                sv[t] = sv[t + s]; si[t] = si[t + s];
            }
        }
        __syncthreads();
    }
    if (t == 0) tok[b] = si[0];
}

extern "C" void kernel_launch(void* const* d_in, const int* in_sizes, int n_in,
                              void* d_out, int out_size, void* d_ws, size_t ws_size,
                              hipStream_t stream) {
    const float* enc = (const float*)d_in[0];   // (64, 1024)
    const int*   tgt = (const int*)  d_in[1];   // (64, 128)
    const float* emb = (const float*)d_in[2];   // (32000, 1024)
    const float* wih = (const float*)d_in[3];   // (3072, 1024)
    const float* whh = (const float*)d_in[4];   // (3072, 1024)
    const float* bih = (const float*)d_in[5];   // (3072,)
    const float* bhh = (const float*)d_in[6];   // (3072,)
    const float* fcw = (const float*)d_in[7];   // (32000, 1024)
    const float* fcb = (const float*)d_in[8];   // (32000,)
    float* out = (float*)d_out;                 // (64, 128, 32000)

    // workspace carving (256-byte aligned)
    char* ws = (char*)d_ws;
    size_t off = 0;
    auto carve = [&](size_t bytes) -> void* {
        off = (off + 255) & ~(size_t)255;
        void* p = ws + off;
        off += bytes;
        return p;
    };
    unsigned short* fcw_p = (unsigned short*)carve((size_t)VOCAB * HID * 2);  // 65.5 MB
    unsigned short* wih_p = (unsigned short*)carve((size_t)3 * HID * HID * 2);
    unsigned short* whh_p = (unsigned short*)carve((size_t)3 * HID * HID * 2);
    unsigned short* x_pk  = (unsigned short*)carve((size_t)BATCH * HID * 2);
    unsigned short* h_pk  = (unsigned short*)carve((size_t)BATCH * HID * 2);
    float*          h     = (float*)carve((size_t)BATCH * HID * 4);
    float*          gi    = (float*)carve((size_t)BATCH * 3 * HID * 4);
    float*          gh    = (float*)carve((size_t)BATCH * 3 * HID * 4);
    int*            tok   = (int*)carve((size_t)BATCH * 4);

    // once per launch: repack weights to bf16 fragment order + state init
    {
        long long n = (long long)VOCAB * HID;
        pack_w_bf16<<<(unsigned)((n + 255) / 256), 256, 0, stream>>>(fcw, fcw_p, n);
        n = (long long)3 * HID * HID;
        pack_w_bf16<<<(unsigned)((n + 255) / 256), 256, 0, stream>>>(wih, wih_p, n);
        pack_w_bf16<<<(unsigned)((n + 255) / 256), 256, 0, stream>>>(whh, whh_p, n);
        init_state<<<(BATCH * HID + 255) / 256, 256, 0, stream>>>(enc, tgt, h, h_pk, tok);
    }

    for (int t = 0; t < TLEN; ++t) {
        embed_cast<<<(BATCH * HID + 255) / 256, 256, 0, stream>>>(emb, tok, x_pk);
        // 96 strips (2 matrices x 48) x 2 M-strips; 4 waves/block
        gates_gemm<<<dim3(24, 2), 128, 0, stream>>>(x_pk, h_pk, wih_p, whh_p, gi, gh);
        gru_update<<<(BATCH * HID + 255) / 256, 256, 0, stream>>>(gi, gh, bih, bhh, h, h_pk);
        float* out_t = out + (size_t)t * VOCAB;
        // 500 N-strips x 2 M-strips
        logits_gemm<<<dim3(125, 2), 128, 0, stream>>>(h_pk, fcw_p, fcb, out_t);
        argmax_rows<<<BATCH, 256, 0, stream>>>(out_t, tok);
    }
}
